// SPConvResBlock_29850022708095
// MI455X (gfx1250) — matrix operand order repaired and checked by reading, not compiled, not run on hardware
//
#include <hip/hip_runtime.h>
#include <hip/hip_bf16.h>
#include <math.h>

// ---------------- problem constants ----------------
static constexpr int Bc   = 4;
static constexpr int IMGc = 256;
static constexpr int Cc   = 128;
static constexpr int Kc   = 7;
static constexpr int TDc  = 512;
static constexpr int ZDc  = 256;
static constexpr int MULTc = 2;
static constexpr int ACTc = IMGc * IMGc / 2;   // 32768
static constexpr int Nc   = Bc * ACTc;         // 131072
static constexpr int PADc = Kc / 2;            // 3
static constexpr int TAPS = Kc * Kc;           // 49
static constexpr int HID  = MULTc * Cc;        // 256

// ---------------- vector types ----------------
typedef __attribute__((ext_vector_type(16))) __bf16        v16bf;
typedef __attribute__((ext_vector_type(8)))  float         v8f;
typedef __attribute__((ext_vector_type(4)))  unsigned int  u32x4;
typedef __attribute__((ext_vector_type(2)))  unsigned int  u32x2;

union FragBF {
    struct { u32x4 lo, hi; } u;   // 32 bytes
    v16bf v;                      // 16 x bf16
};

__device__ inline v8f zero_v8f() {
    v8f z;
#pragma unroll
    for (int i = 0; i < 8; ++i) z[i] = 0.0f;
    return z;
}

// round-to-nearest-even float -> bf16 (as raw ushort)
__device__ inline unsigned short f2bf(float f) {
    unsigned int u = __float_as_uint(f);
    u += 0x7FFFu + ((u >> 16) & 1u);
    return (unsigned short)(u >> 16);
}

__device__ inline float gelu_exact(float x) {
    return 0.5f * x * (1.0f + erff(x * 0.70710678118654752f));
}

// ---------------- prep kernels ----------------
__global__ void grid_init_kernel(int* __restrict__ g) {
    int i = blockIdx.x * 256 + threadIdx.x;
    if (i < Bc * IMGc * IMGc) g[i] = -1;
}

__global__ void zpage_kernel(unsigned int* __restrict__ zp) {
    zp[threadIdx.x] = 0u;                    // <<<1,64>>> -> 256 zero bytes
}

__global__ void grid_scatter_kernel(const int* __restrict__ xidx, int* __restrict__ g) {
    int i = blockIdx.x * 256 + threadIdx.x;
    if (i >= Nc) return;
    int b = xidx[i * 3 + 0], y = xidx[i * 3 + 1], x = xidx[i * 3 + 2];
    g[(b * IMGc + y) * IMGc + x] = i;
}

// conv_w [tap][cin][cout] f32 -> wt [tap][cout][cin] bf16
__global__ void prep_convw_kernel(const float* __restrict__ cw, unsigned short* __restrict__ wt) {
    int i = blockIdx.x * 256 + threadIdx.x;
    if (i >= TAPS * Cc * Cc) return;
    int tap = i >> 14;
    int rem = i & 16383;
    int co  = rem >> 7;
    int ci  = rem & 127;
    wt[i] = f2bf(cw[(size_t)tap * Cc * Cc + ci * Cc + co]);
}

// mlp_w1 [128][256] -> wt1 [256][128] bf16 (B fragment wants [n][k])
__global__ void prep_w1_kernel(const float* __restrict__ w, unsigned short* __restrict__ wt) {
    int i = blockIdx.x * 256 + threadIdx.x;
    if (i >= HID * Cc) return;
    int n = i >> 7, k = i & 127;
    wt[i] = f2bf(w[k * HID + n]);
}

// mlp_w2 [256][128] -> wt2 [128][256] bf16
__global__ void prep_w2_kernel(const float* __restrict__ w, unsigned short* __restrict__ wt) {
    int i = blockIdx.x * 256 + threadIdx.x;
    if (i >= Cc * HID) return;
    int n = i >> 8, k = i & 255;
    wt[i] = f2bf(w[k * Cc + n]);
}

// per-batch modulation vectors: tt = gelu(t) @ tw + tb (256);
// zz = gelu(z @ zw1 + zb1) @ zw2 + zb2 (128).  blockIdx.x = b*2 + pass.
__global__ void modvec_kernel(const float* __restrict__ t, const float* __restrict__ z,
                              const float* __restrict__ t1w, const float* __restrict__ t1b,
                              const float* __restrict__ t2w, const float* __restrict__ t2b,
                              const float* __restrict__ z1w1, const float* __restrict__ z1b1,
                              const float* __restrict__ z1w2, const float* __restrict__ z1b2,
                              const float* __restrict__ z2w1, const float* __restrict__ z2b1,
                              const float* __restrict__ z2w2, const float* __restrict__ z2b2,
                              float* __restrict__ tt1, float* __restrict__ zz1,
                              float* __restrict__ tt2, float* __restrict__ zz2) {
    int b = blockIdx.x >> 1, p = blockIdx.x & 1;
    const float* tw  = p ? t2w  : t1w;
    const float* tb  = p ? t2b  : t1b;
    const float* zw1 = p ? z2w1 : z1w1;
    const float* zb1 = p ? z2b1 : z1b1;
    const float* zw2 = p ? z2w2 : z1w2;
    const float* zb2 = p ? z2b2 : z1b2;
    float* tt = p ? tt2 : tt1;
    float* zz = p ? zz2 : zz1;

    __shared__ float gt[TDc];
    __shared__ float zh[Cc];
    int tid = threadIdx.x;

    for (int i = tid; i < TDc; i += 256) gt[i] = gelu_exact(t[b * TDc + i]);
    __syncthreads();

    {   // tt[tid], tid in [0,256)
        float s = tb[tid];
        for (int k = 0; k < TDc; ++k) s += gt[k] * tw[k * (2 * Cc) + tid];
        tt[b * (2 * Cc) + tid] = s;
    }
    if (tid < Cc) {
        float s = zb1[tid];
        for (int k = 0; k < ZDc; ++k) s += z[b * ZDc + k] * zw1[k * Cc + tid];
        zh[tid] = gelu_exact(s);
    }
    __syncthreads();
    if (tid < Cc) {
        float s = zb2[tid];
        for (int k = 0; k < Cc; ++k) s += zh[k] * zw2[k * Cc + tid];
        zz[b * Cc + tid] = s;
    }
}

// ---------------- modulate pass 1: LN + t/z modulation -> bf16 ----------------
// one wave (32 lanes) per row, 4 channels/lane; 8 rows per 256-thread block.
__global__ void modulate1_kernel(const float* __restrict__ xf,
                                 const float* __restrict__ lnw, const float* __restrict__ lnb,
                                 const float* __restrict__ tt,  const float* __restrict__ zz,
                                 unsigned short* __restrict__ mod1) {
    int tid = threadIdx.x, wave = tid >> 5, lane = tid & 31;
    int row = blockIdx.x * 8 + wave;
    int b = row >> 15;                       // row / ACT
    const float* xr = xf + (size_t)row * Cc + lane * 4;
    float v0 = xr[0], v1 = xr[1], v2 = xr[2], v3 = xr[3];

    float s = v0 + v1 + v2 + v3;
#pragma unroll
    for (int off = 16; off >= 1; off >>= 1) s += __shfl_xor(s, off, 32);
    float mu = s * (1.0f / 128.0f);
    float d0 = v0 - mu, d1 = v1 - mu, d2 = v2 - mu, d3 = v3 - mu;
    float q = d0 * d0 + d1 * d1 + d2 * d2 + d3 * d3;
#pragma unroll
    for (int off = 16; off >= 1; off >>= 1) q += __shfl_xor(q, off, 32);
    float rs = rsqrtf(q * (1.0f / 128.0f) + 1e-5f);

    int c = lane * 4;
    float d[4] = { d0, d1, d2, d3 };
    unsigned int pk0 = 0, pk1 = 0;
#pragma unroll
    for (int j = 0; j < 4; ++j) {
        float g = d[j] * rs * lnw[c + j] + lnb[c + j];
        g = g * (1.0f + tt[b * 256 + c + j]) + tt[b * 256 + 128 + c + j];
        g = g * (1.0f + zz[b * 128 + c + j]);
        unsigned int h = f2bf(g);
        if (j == 0) pk0 = h;
        else if (j == 1) pk0 |= h << 16;
        else if (j == 2) pk1 = h;
        else pk1 |= h << 16;
    }
    u32x2 out; out[0] = pk0; out[1] = pk1;
    *(u32x2*)(mod1 + (size_t)row * Cc + c) = out;
}

// ---------------- gather conv: WMMA + double-buffered async->LDS gather ------
// block = 64 rows x 128 cout; 8 waves; wave w owns n0 = 16w for 4 m-subtiles.
// Pipeline: async DMA for tap+1 runs while WMMAs for tap execute; the partial
// wait (asynccnt<=4) exploits in-order completion of async loads per wave.
__global__ void conv_wmma_kernel(const float* __restrict__ xfeat,
                                 const int* __restrict__ xidx,
                                 const float* __restrict__ norm,
                                 const unsigned short* __restrict__ mod1,
                                 const unsigned short* __restrict__ wt,
                                 const int* __restrict__ grid,
                                 const unsigned short* __restrict__ zpage,
                                 float* __restrict__ xout) {
    __shared__ __align__(16) unsigned short As[2][64 * Cc];   // 2 x 16 KB A tiles
    __shared__ int s_b[64], s_y[64], s_x[64];
    __shared__ unsigned long long s_ga[2][64];                // per-row gather src addr

    int m0  = blockIdx.x * 64;
    int tid = threadIdx.x;
    int wave = tid >> 5, lane = tid & 31;
    int n0 = wave * 16;

    if (tid < 64) {
        int r = m0 + tid;
        s_b[tid] = xidx[r * 3 + 0];
        s_y[tid] = xidx[r * 3 + 1];
        s_x[tid] = xidx[r * 3 + 2];
    }
    __syncthreads();

    // neighbor source address for a tap (valid for tid < 64)
    auto tap_addr = [&](int tap_i) -> unsigned long long {
        int dy = tap_i / Kc - PADc, dx = tap_i % Kc - PADc;
        int ny = s_y[tid] + dy, nx = s_x[tid] + dx;
        int g = -1;
        if (ny >= 0 && ny < IMGc && nx >= 0 && nx < IMGc)
            g = grid[(s_b[tid] * IMGc + ny) * IMGc + nx];
        return (g >= 0) ? (unsigned long long)(mod1 + (size_t)g * Cc)
                        : (unsigned long long)zpage;
    };
    // 64 rows x 256B, 16B per issue, 4 issues/thread, direct to LDS (ASYNCcnt)
    auto issue_gather = [&](int bufi) {
#pragma unroll
        for (int i = 0; i < 4; ++i) {
            int id  = tid + 256 * i;
            int row = id >> 4, ch = id & 15;
            unsigned long long ga = s_ga[bufi][row] + (unsigned long long)(ch * 16);
            unsigned int la = (unsigned int)(unsigned long long)&As[bufi][row * Cc + ch * 8];
            asm volatile("global_load_async_to_lds_b128 %0, %1, off"
                         :: "v"(la), "v"(ga) : "memory");
        }
    };

    v8f acc[4] = { zero_v8f(), zero_v8f(), zero_v8f(), zero_v8f() };

    // prologue: addresses for tap0 -> issue; precompute addresses for tap1
    if (tid < 64) s_ga[0][tid] = tap_addr(0);
    __syncthreads();
    issue_gather(0);
    if (tid < 64) s_ga[1][tid] = tap_addr(1);

    for (int tap = 0; tap < TAPS; ++tap) {
        int cur = tap & 1, nxt = cur ^ 1;
        __syncthreads();                 // (A) s_ga[nxt] ready; buf[nxt] free
        if (tap + 1 < TAPS) {
            issue_gather(nxt);           // tap+1 DMA overlaps with tap's WMMAs
            asm volatile("s_wait_asynccnt 0x4" ::: "memory");   // tap's 4 done
        } else {
            asm volatile("s_wait_asynccnt 0x0" ::: "memory");
        }
        __syncthreads();                 // (B) buf[cur] visible to all waves

        const unsigned short* wb = wt + (size_t)tap * Cc * Cc;   // [cout][cin] bf16
        const unsigned short* Ab = &As[cur][0];
#pragma unroll
        for (int kc = 0; kc < 4; ++kc) {
            int k0 = kc * 32;
            FragBF bmat;
            int brow  = n0 + (lane & 15);
            int koffB = k0 + ((lane >> 4) << 3);                 // +0 or +8
            bmat.u.lo = *(const u32x4*)(wb + (size_t)brow * Cc + koffB);
            bmat.u.hi = *(const u32x4*)(wb + (size_t)brow * Cc + koffB + 16);
            int koffA = k0 + ((lane >> 4) << 3);                 // +0 or +8
#pragma unroll
            for (int mt = 0; mt < 4; ++mt) {
                FragBF a;
                int arow = mt * 16 + (lane & 15);
                a.u.lo = *(const u32x4*)(Ab + arow * Cc + koffA);
                a.u.hi = *(const u32x4*)(Ab + arow * Cc + koffA + 16);
                acc[mt] = __builtin_amdgcn_wmma_f32_16x16x32_bf16(
                    false, a.v, false, bmat.v, (short)0, acc[mt], false, false);
            }
        }

        // addresses for tap+2 into s_ga[cur]; its last readers were pre-(A)
        if (tap + 2 < TAPS && tid < 64) s_ga[cur][tid] = tap_addr(tap + 2);
    }

    // epilogue: x = x_features + conv/norm
#pragma unroll
    for (int mt = 0; mt < 4; ++mt) {
#pragma unroll
        for (int r = 0; r < 8; ++r) {
            int rowl = mt * 16 + r + ((lane >> 4) << 3);
            int row  = m0 + rowl;
            int col  = n0 + (lane & 15);
            xout[(size_t)row * Cc + col] =
                xfeat[(size_t)row * Cc + col] + acc[mt][r] / norm[row];
        }
    }
}

// ---------------- fused modulate2 + MLP + residual (32 rows / block) --------
__global__ void mlp_wmma_kernel(const float* __restrict__ xbuf,
                                const float* __restrict__ lnw, const float* __restrict__ lnb,
                                const float* __restrict__ tt,  const float* __restrict__ zz,
                                const unsigned short* __restrict__ wt1,
                                const unsigned short* __restrict__ wt2,
                                const float* __restrict__ b1, const float* __restrict__ b2,
                                float* __restrict__ out) {
    __shared__ __align__(16) unsigned short Am[32 * Cc];    //  8 KB
    __shared__ __align__(16) unsigned short Hs[32 * HID];   // 16 KB
    __shared__ float Xs[32 * Cc];                           // 16 KB (residual)

    int m0  = blockIdx.x * 32;
    int tid = threadIdx.x, wave = tid >> 5, lane = tid & 31;

    // stage 0: LN + modulate (4 rows per wave)
#pragma unroll
    for (int rr = 0; rr < 4; ++rr) {
        int rowl = wave * 4 + rr;
        int row  = m0 + rowl;
        int b    = row >> 15;
        const float* xr = xbuf + (size_t)row * Cc + lane * 4;
        float v0 = xr[0], v1 = xr[1], v2 = xr[2], v3 = xr[3];
        float s = v0 + v1 + v2 + v3;
#pragma unroll
        for (int off = 16; off >= 1; off >>= 1) s += __shfl_xor(s, off, 32);
        float mu = s * (1.0f / 128.0f);
        float d0 = v0 - mu, d1 = v1 - mu, d2 = v2 - mu, d3 = v3 - mu;
        float q = d0 * d0 + d1 * d1 + d2 * d2 + d3 * d3;
#pragma unroll
        for (int off = 16; off >= 1; off >>= 1) q += __shfl_xor(q, off, 32);
        float rs = rsqrtf(q * (1.0f / 128.0f) + 1e-5f);
        float v[4] = { v0, v1, v2, v3 };
        float d[4] = { d0, d1, d2, d3 };
        int c = lane * 4;
#pragma unroll
        for (int j = 0; j < 4; ++j) {
            float g = d[j] * rs * lnw[c + j] + lnb[c + j];
            g = g * (1.0f + tt[b * 256 + c + j]) + tt[b * 256 + 128 + c + j];
            g = g * (1.0f + zz[b * 128 + c + j]);
            Am[rowl * Cc + c + j] = f2bf(g);
            Xs[rowl * Cc + c + j] = v[j];
        }
    }
    __syncthreads();

    // GEMM1: hidden = Am(32x128) @ W1(128x256); wave owns 2 n-tiles x 2 m-tiles
    v8f acc1[2][2] = { { zero_v8f(), zero_v8f() }, { zero_v8f(), zero_v8f() } };
#pragma unroll
    for (int kc = 0; kc < 4; ++kc) {
        int k0 = kc * 32;
        int koffA = k0 + ((lane >> 4) << 3);
        int koffB = k0 + ((lane >> 4) << 3);
#pragma unroll
        for (int ts = 0; ts < 2; ++ts) {
            int brow = wave * 32 + ts * 16 + (lane & 15);
            FragBF bmat;
            bmat.u.lo = *(const u32x4*)(wt1 + (size_t)brow * Cc + koffB);
            bmat.u.hi = *(const u32x4*)(wt1 + (size_t)brow * Cc + koffB + 16);
#pragma unroll
            for (int mt = 0; mt < 2; ++mt) {
                FragBF a;
                int arow = mt * 16 + (lane & 15);
                a.u.lo = *(const u32x4*)(&Am[arow * Cc + koffA]);
                a.u.hi = *(const u32x4*)(&Am[arow * Cc + koffA + 16]);
                acc1[mt][ts] = __builtin_amdgcn_wmma_f32_16x16x32_bf16(
                    false, a.v, false, bmat.v, (short)0, acc1[mt][ts], false, false);
            }
        }
    }
    // bias + exact GELU -> bf16 hidden tile
#pragma unroll
    for (int ts = 0; ts < 2; ++ts) {
#pragma unroll
        for (int mt = 0; mt < 2; ++mt) {
#pragma unroll
            for (int r = 0; r < 8; ++r) {
                int rowl = mt * 16 + r + ((lane >> 4) << 3);
                int col  = wave * 32 + ts * 16 + (lane & 15);
                float h = acc1[mt][ts][r] + b1[col];
                Hs[rowl * HID + col] = f2bf(gelu_exact(h));
            }
        }
    }
    __syncthreads();

    // GEMM2: out_h = Hs(32x256) @ W2(256x128); wave owns n0 = 16*wave, 2 m-tiles
    v8f acc2[2] = { zero_v8f(), zero_v8f() };
    int n0 = wave * 16;
#pragma unroll
    for (int kc = 0; kc < 8; ++kc) {
        int k0 = kc * 32;
        FragBF bmat;
        int brow  = n0 + (lane & 15);
        int koffB = k0 + ((lane >> 4) << 3);
        bmat.u.lo = *(const u32x4*)(wt2 + (size_t)brow * HID + koffB);
        bmat.u.hi = *(const u32x4*)(wt2 + (size_t)brow * HID + koffB + 16);
        int koffA = k0 + ((lane >> 4) << 3);
#pragma unroll
        for (int mt = 0; mt < 2; ++mt) {
            FragBF a;
            int arow = mt * 16 + (lane & 15);
            a.u.lo = *(const u32x4*)(&Hs[arow * HID + koffA]);
            a.u.hi = *(const u32x4*)(&Hs[arow * HID + koffA + 16]);
            acc2[mt] = __builtin_amdgcn_wmma_f32_16x16x32_bf16(
                false, a.v, false, bmat.v, (short)0, acc2[mt], false, false);
        }
    }
    // residual + bias2 -> final output (in-place over xbuf == out is safe:
    // this block only reads/writes its own 32 rows; reads happened in stage 0)
#pragma unroll
    for (int mt = 0; mt < 2; ++mt) {
#pragma unroll
        for (int r = 0; r < 8; ++r) {
            int rowl = mt * 16 + r + ((lane >> 4) << 3);
            int col  = n0 + (lane & 15);
            out[(size_t)(m0 + rowl) * Cc + col] = Xs[rowl * Cc + col] + acc2[mt][r] + b2[col];
        }
    }
}

// ---------------- launcher ----------------
extern "C" void kernel_launch(void* const* d_in, const int* in_sizes, int n_in,
                              void* d_out, int out_size, void* d_ws, size_t ws_size,
                              hipStream_t stream) {
    (void)in_sizes; (void)n_in; (void)out_size; (void)ws_size;

    const float* x_features = (const float*)d_in[0];
    const int*   x_indices  = (const int*)  d_in[1];
    const float* t          = (const float*)d_in[2];
    const float* z          = (const float*)d_in[3];
    const float* norm       = (const float*)d_in[4];
    const float* ln1w       = (const float*)d_in[5];
    const float* ln1b       = (const float*)d_in[6];
    const float* ln2w       = (const float*)d_in[7];
    const float* ln2b       = (const float*)d_in[8];
    const float* conv_w     = (const float*)d_in[9];
    const float* t1w        = (const float*)d_in[10];
    const float* t1b        = (const float*)d_in[11];
    const float* t2w        = (const float*)d_in[12];
    const float* t2b        = (const float*)d_in[13];
    const float* z1w1       = (const float*)d_in[14];
    const float* z1b1       = (const float*)d_in[15];
    const float* z1w2       = (const float*)d_in[16];
    const float* z1b2       = (const float*)d_in[17];
    const float* z2w1       = (const float*)d_in[18];
    const float* z2b1       = (const float*)d_in[19];
    const float* z2w2       = (const float*)d_in[20];
    const float* z2b2       = (const float*)d_in[21];
    const float* mlp_w1     = (const float*)d_in[22];
    const float* mlp_b1     = (const float*)d_in[23];
    const float* mlp_w2     = (const float*)d_in[24];
    const float* mlp_b2     = (const float*)d_in[25];

    // ---- workspace carve (256B aligned) ----
    char* wp = (char*)d_ws;
    auto carve = [&](size_t bytes) -> char* {
        char* p = wp;
        wp += (bytes + 255) & ~(size_t)255;
        return p;
    };
    int*            grid  = (int*)           carve(sizeof(int) * Bc * IMGc * IMGc);   // 1 MB
    unsigned int*   zpage = (unsigned int*)  carve(256);                              // zero page
    float*          tt1   = (float*)         carve(sizeof(float) * Bc * 2 * Cc);
    float*          zz1   = (float*)         carve(sizeof(float) * Bc * Cc);
    float*          tt2   = (float*)         carve(sizeof(float) * Bc * 2 * Cc);
    float*          zz2   = (float*)         carve(sizeof(float) * Bc * Cc);
    unsigned short* wtc   = (unsigned short*)carve(sizeof(short) * TAPS * Cc * Cc);   // 1.57 MB
    unsigned short* wt1   = (unsigned short*)carve(sizeof(short) * HID * Cc);
    unsigned short* wt2   = (unsigned short*)carve(sizeof(short) * Cc * HID);
    unsigned short* mod1  = (unsigned short*)carve(sizeof(short) * (size_t)Nc * Cc);  // 33.5 MB

    float* xbuf = (float*)d_out;   // post-conv residual lives in d_out; mlp finalizes in place

    // ---- prep ----
    grid_init_kernel<<<(Bc * IMGc * IMGc + 255) / 256, 256, 0, stream>>>(grid);
    zpage_kernel<<<1, 64, 0, stream>>>(zpage);
    grid_scatter_kernel<<<(Nc + 255) / 256, 256, 0, stream>>>(x_indices, grid);
    prep_convw_kernel<<<(TAPS * Cc * Cc + 255) / 256, 256, 0, stream>>>(conv_w, wtc);
    prep_w1_kernel<<<(HID * Cc + 255) / 256, 256, 0, stream>>>(mlp_w1, wt1);
    prep_w2_kernel<<<(Cc * HID + 255) / 256, 256, 0, stream>>>(mlp_w2, wt2);
    modvec_kernel<<<Bc * 2, 256, 0, stream>>>(t, z,
        t1w, t1b, t2w, t2b,
        z1w1, z1b1, z1w2, z1b2,
        z2w1, z2b1, z2w2, z2b2,
        tt1, zz1, tt2, zz2);

    // ---- main pipeline ----
    modulate1_kernel<<<Nc / 8, 256, 0, stream>>>(x_features, ln1w, ln1b, tt1, zz1, mod1);
    conv_wmma_kernel<<<Nc / 64, 256, 0, stream>>>(x_features, x_indices, norm,
                                                  mod1, wtc, grid, (const unsigned short*)zpage,
                                                  xbuf);
    mlp_wmma_kernel<<<Nc / 32, 256, 0, stream>>>(xbuf, ln2w, ln2b, tt2, zz2,
                                                 wt1, wt2, mlp_b1, mlp_b2, (float*)d_out);
}
